// MiMoV2FlashAttention_7679401525651
// MI455X (gfx1250) — compile-verified, matrix operand order checked by reading, not run on hardware
//
#include <hip/hip_runtime.h>
#include <hip/hip_bf16.h>

typedef __attribute__((ext_vector_type(16))) __bf16 v16bf;
typedef __attribute__((ext_vector_type(8)))  float  v8f;
typedef int          v4i  __attribute__((__vector_size__(4 * sizeof(int))));
typedef unsigned int v4u  __attribute__((__vector_size__(4 * sizeof(int))));
typedef int          v8i32 __attribute__((__vector_size__(8 * sizeof(int))));

#define HIDDEN 4096
#define NH     32
#define NKV    8
#define HD     128
#define VHD    96
#define ROT    64
#define BATCH  2
#define SEQ    2048
#define BS     (BATCH * SEQ)   /* 4096 */

// ---- CDNA5 async global->LDS path (guarded; falls back to sync copy) ------
#if defined(__HIP_DEVICE_COMPILE__) && \
    __has_builtin(__builtin_amdgcn_global_load_async_to_lds_b128) && \
    __has_builtin(__builtin_amdgcn_s_wait_asynccnt)
#define USE_ASYNC_LDS 1
#else
#define USE_ASYNC_LDS 0
#endif

// ---- CDNA5 Tensor Data Mover path (guarded) -------------------------------
#if defined(__HIP_DEVICE_COMPILE__) && \
    __has_builtin(__builtin_amdgcn_tensor_load_to_lds) && \
    __has_builtin(__builtin_amdgcn_s_wait_tensorcnt)
#define USE_TDM 1
#else
#define USE_TDM 0
#endif

__device__ __forceinline__ void cp_async16(unsigned short* dst_lds,
                                           const unsigned short* src_g) {
#if USE_ASYNC_LDS
  __builtin_amdgcn_global_load_async_to_lds_b128(
      (__attribute__((address_space(1))) v4i*)src_g,
      (__attribute__((address_space(3))) v4i*)dst_lds, 0, 0);
#else
  *reinterpret_cast<int4*>(dst_lds) = *reinterpret_cast<const int4*>(src_g);
#endif
}
__device__ __forceinline__ void async_wait_all() {
#if USE_ASYNC_LDS
  __builtin_amdgcn_s_wait_asynccnt(0);
#endif
}

// DMA a 2D tile (tile_h rows x tile_w bf16 elements, row stride in elements)
// from global into LDS (row-major, contiguous) via the Tensor Data Mover.
__device__ __forceinline__ void tdm_load_2d(unsigned short* lds_dst,
                                            const unsigned short* gsrc,
                                            unsigned tile_w, unsigned tile_h,
                                            unsigned row_stride) {
#if USE_TDM
  unsigned long long ga = (unsigned long long)(size_t)gsrc;
  unsigned ldsa =
      (unsigned)(size_t)((__attribute__((address_space(3))) unsigned short*)lds_dst);
  v4u g0;
  g0[0] = 1u;                                     // count=1 (valid user D#)
  g0[1] = ldsa;                                   // lds_addr (bytes)
  g0[2] = (unsigned)(ga & 0xFFFFFFFFu);           // global_addr[31:0]
  g0[3] = (unsigned)((ga >> 32) & 0x01FFFFFFu)    // global_addr[56:32]
        | (2u << 30);                             // type = 2 ("image")
  v8i32 g1;
  g1[0] = (int)(1u << 16);                        // data_size=1 -> 2-byte elems
  g1[1] = (int)((tile_w & 0xFFFFu) << 16);        // tensor_dim0[15:0]
  g1[2] = (int)(((tile_w >> 16) & 0xFFFFu)        // tensor_dim0[31:16]
        | ((tile_h & 0xFFFFu) << 16));            // tensor_dim1[15:0]
  g1[3] = (int)(((tile_h >> 16) & 0xFFFFu)        // tensor_dim1[31:16]
        | ((tile_w & 0xFFFFu) << 16));            // tile_dim0
  g1[4] = (int)(tile_h & 0xFFFFu);                // tile_dim1 (tile_dim2=0)
  g1[5] = (int)row_stride;                        // tensor_dim0_stride[31:0]
  g1[6] = 0;                                      // stride[47:32] | dim1_stride lo
  g1[7] = 0;
  v4i z = {0, 0, 0, 0};
#if __clang_major__ >= 23
  v8i32 z8 = {};
  __builtin_amdgcn_tensor_load_to_lds(g0, g1, z, z, z8, 0);
#else
  __builtin_amdgcn_tensor_load_to_lds(g0, g1, z, z, 0);
#endif
#else
  (void)lds_dst; (void)gsrc; (void)tile_w; (void)tile_h; (void)row_stride;
#endif
}
__device__ __forceinline__ void tdm_wait_all() {
#if USE_TDM
  __builtin_amdgcn_s_wait_tensorcnt(0);
#endif
}

__device__ __forceinline__ unsigned short f2bf(float f) {
  unsigned int u = __float_as_uint(f);
  u += 0x7FFFu + ((u >> 16) & 1u);           // round-to-nearest-even
  return (unsigned short)(u >> 16);
}
__device__ __forceinline__ float bf2f(unsigned short h) {
  return __uint_as_float(((unsigned int)h) << 16);
}

// ---------------------------------------------------------------------------
// fp32 -> bf16 conversion (vectorized, float4 in / ushort4 out)
// ---------------------------------------------------------------------------
__global__ __launch_bounds__(256) void cvt_f32_bf16(
    const float* __restrict__ src, unsigned short* __restrict__ dst, int n4) {
  int i = blockIdx.x * 256 + threadIdx.x;
  if (i >= n4) return;
  float4 f = reinterpret_cast<const float4*>(src)[i];
  ushort4 o;
  o.x = f2bf(f.x); o.y = f2bf(f.y); o.z = f2bf(f.z); o.w = f2bf(f.w);
  reinterpret_cast<ushort4*>(dst)[i] = o;
}

// ---------------------------------------------------------------------------
// C[m,n] = sum_k A[m,k] * W[n,k]   (A:[M,K] bf16, W:[N,K] bf16, row-major)
// Workgroup tile 128x128, 8 waves, each wave owns 32x64 (2x4 WMMA tiles).
// Double-buffered LDS staging via async global->LDS loads.
// ---------------------------------------------------------------------------
template <bool F32OUT>
__global__ __launch_bounds__(256) void gemm_bf16_wmma(
    const unsigned short* __restrict__ A,
    const unsigned short* __restrict__ W,
    void* __restrict__ Cout, int M, int N, int K) {
  __shared__ __align__(32) unsigned short sA[2][128 * 32];
  __shared__ __align__(32) unsigned short sW[2][128 * 32];
  const int tid  = threadIdx.x;
  const int lane = tid & 31;
  const int wave = tid >> 5;
  const int m0 = blockIdx.y * 128;
  const int n0 = blockIdx.x * 128;
  const int wm = (wave >> 1) * 32;      // 0,32,64,96
  const int wn = (wave & 1) * 64;       // 0,64
  const int lr      = lane & 15;
  const int halfk   = (lane >> 4) << 4; // 0 or 16 (K-half per lane group)
  const int rowHalf = (lane >> 4) << 3; // 0 or 8  (C row offset per lane group)
  v8f acc[2][4] = {};
  const int kTiles = K >> 5;

  auto stage = [&](int buf, int kt) {
    const int kb = kt << 5;
#pragma unroll
    for (int i = 0; i < 2; ++i) {       // A tile: 128x32 = 512 x 16B
      int le = tid + (i << 8);
      int r = le >> 2, c = (le & 3) << 3;
      cp_async16(&sA[buf][r * 32 + c], &A[(size_t)(m0 + r) * K + kb + c]);
    }
#pragma unroll
    for (int i = 0; i < 2; ++i) {       // W tile: 128x32 = 512 x 16B
      int le = tid + (i << 8);
      int r = le >> 2, c = (le & 3) << 3;
      cp_async16(&sW[buf][r * 32 + c], &W[(size_t)(n0 + r) * K + kb + c]);
    }
  };

  stage(0, 0);
  async_wait_all();
  __syncthreads();

  for (int kt = 0; kt < kTiles; ++kt) {
    const int cur = kt & 1;
    if (kt + 1 < kTiles) stage(cur ^ 1, kt + 1);   // prefetch next stage
    v16bf af[2], bw[4];
#pragma unroll
    for (int i = 0; i < 2; ++i)
      af[i] = *reinterpret_cast<const v16bf*>(&sA[cur][(wm + i * 16 + lr) * 32 + halfk]);
#pragma unroll
    for (int j = 0; j < 4; ++j)
      bw[j] = *reinterpret_cast<const v16bf*>(&sW[cur][(wn + j * 16 + lr) * 32 + halfk]);
#pragma unroll
    for (int i = 0; i < 2; ++i)
#pragma unroll
      for (int j = 0; j < 4; ++j)
        acc[i][j] = __builtin_amdgcn_wmma_f32_16x16x32_bf16(
            false, af[i], false, bw[j], (short)0, acc[i][j], false, false);
    async_wait_all();
    __syncthreads();
  }
#pragma unroll
  for (int i = 0; i < 2; ++i)
#pragma unroll
    for (int j = 0; j < 4; ++j)
#pragma unroll
      for (int e = 0; e < 8; ++e) {
        int m = m0 + wm + i * 16 + e + rowHalf;
        int n = n0 + wn + j * 16 + lr;
        if constexpr (F32OUT)
          reinterpret_cast<float*>(Cout)[(size_t)m * N + n] = acc[i][j][e];
        else
          reinterpret_cast<unsigned short*>(Cout)[(size_t)m * N + n] = f2bf(acc[i][j][e]);
      }
}

// ---------------------------------------------------------------------------
// RoPE on first ROT dims of every Q/K head, in place on bf16.
// ---------------------------------------------------------------------------
__global__ __launch_bounds__(256) void rope_kernel(
    unsigned short* __restrict__ Q, unsigned short* __restrict__ K,
    const int* __restrict__ pos_ids) {
  int idx = blockIdx.x * 256 + threadIdx.x;
  if (idx >= BS * (ROT / 2)) return;
  int i = idx & 31;                 // pair index 0..31
  int t = idx >> 5;                 // flat token (b*S + s)
  float p   = (float)pos_ids[t];
  float inv = __expf(-(float)(2 * i) * (9.210340371976184f / (float)ROT));
  float sn, cs;
  __sincosf(p * inv, &sn, &cs);
  unsigned short* q = Q + (size_t)t * NH * HD;
#pragma unroll 4
  for (int h = 0; h < NH; ++h) {
    float x1 = bf2f(q[h * HD + i]), x2 = bf2f(q[h * HD + i + 32]);
    q[h * HD + i]      = f2bf(x1 * cs - x2 * sn);
    q[h * HD + i + 32] = f2bf(x2 * cs + x1 * sn);
  }
  unsigned short* k = K + (size_t)t * NKV * HD;
#pragma unroll
  for (int h = 0; h < NKV; ++h) {
    float x1 = bf2f(k[h * HD + i]), x2 = bf2f(k[h * HD + i + 32]);
    k[h * HD + i]      = f2bf(x1 * cs - x2 * sn);
    k[h * HD + i + 32] = f2bf(x2 * cs + x1 * sn);
  }
}

// ---------------------------------------------------------------------------
// Fused causal flash attention (GQA), bf16 WMMA, online softmax.
// Block = 128 threads = 4 waves; each wave owns 16 query rows of a 64-row tile.
// grid = (S/64, NH, B).  Q/K tiles staged with the Tensor Data Mover (TDM);
// falls back to async global->LDS per-lane staging.
// ---------------------------------------------------------------------------
__global__ __launch_bounds__(128) void flash_attn_kernel(
    const unsigned short* __restrict__ Q,   // [B,S,NH,HD]
    const unsigned short* __restrict__ K,   // [B,S,NKV,HD]
    const unsigned short* __restrict__ V,   // [B,S,NKV,VHD]
    unsigned short* __restrict__ AO) {      // [B,S,NH*VHD]
  __shared__ __align__(32) unsigned short sQ[64 * HD];    // 16 KB
  __shared__ __align__(32) unsigned short sK[64 * HD];    // 16 KB
  __shared__ __align__(32) unsigned short sVt[VHD * 64];  // 12 KB (transposed)
  __shared__ __align__(32) unsigned short sP[64 * 64];    //  8 KB
  const int tid  = threadIdx.x;
  const int lane = tid & 31;
  const int wave = tid >> 5;
  const int q0 = blockIdx.x * 64;
  const int h  = blockIdx.y;
  const int b  = blockIdx.z;
  const int kh = h / (NH / NKV);
  const int r0 = wave * 16;
  const int lr      = lane & 15;
  const int halfk   = (lane >> 4) << 4;
  const int rowHalf = (lane >> 4) << 3;
  const float scale = 0.08838834764831845f;   // 1/sqrt(128)

  // stage Q tile (64 x 128, row stride NH*HD)
#if USE_TDM
  if (wave == 0)
    tdm_load_2d(sQ, &Q[((size_t)(b * SEQ + q0) * NH + h) * HD], HD, 64, NH * HD);
#else
#pragma unroll
  for (int i = 0; i < 8; ++i) {
    int le = tid + i * 128;
    int r = le >> 4, c = (le & 15) << 3;
    cp_async16(&sQ[r * HD + c],
               &Q[((size_t)(b * SEQ + q0 + r) * NH + h) * HD + c]);
  }
#endif
  float rm[8], rl[8];
#pragma unroll
  for (int e = 0; e < 8; ++e) { rm[e] = -1e30f; rl[e] = 0.0f; }
  v8f o[6] = {};

  const int nkb = blockIdx.x + 1;   // causal: key blocks 0..q-tile
  for (int kb = 0; kb < nkb; ++kb) {
    __syncthreads();
    // stage K tile 64x128 (row stride NKV*HD)
#if USE_TDM
    if (wave == 0)
      tdm_load_2d(sK, &K[((size_t)(b * SEQ + kb * 64) * NKV + kh) * HD], HD, 64, NKV * HD);
#else
#pragma unroll
    for (int i = 0; i < 8; ++i) {
      int le = tid + i * 128;
      int r = le >> 4, c = (le & 15) << 3;
      cp_async16(&sK[r * HD + c],
                 &K[((size_t)(b * SEQ + kb * 64 + r) * NKV + kh) * HD + c]);
    }
#endif
    // stage V tile 64x96, transposed into sVt[vdim][key] (needs shuffle)
#pragma unroll
    for (int i = 0; i < 6; ++i) {
      int le = tid + i * 128;                 // [0,768)
      int r = le / 12, c = (le % 12) * 8;
      __align__(16) unsigned short tmp[8];
      *reinterpret_cast<int4*>(tmp) =
          *reinterpret_cast<const int4*>(&V[((size_t)(b * SEQ + kb * 64 + r) * NKV + kh) * VHD + c]);
#pragma unroll
      for (int j = 0; j < 8; ++j) sVt[(c + j) * 64 + r] = tmp[j];
    }
#if USE_TDM
    if (wave == 0) tdm_wait_all();
#else
    async_wait_all();
#endif
    __syncthreads();

    // S = Q * K^T  (16 rows x 64 keys per wave; contract over d=128)
    v8f sacc[4] = {};
#pragma unroll
    for (int t = 0; t < 4; ++t)
#pragma unroll
      for (int ks = 0; ks < 4; ++ks) {
        v16bf a  = *reinterpret_cast<const v16bf*>(&sQ[(r0 + lr) * HD + ks * 32 + halfk]);
        v16bf bb = *reinterpret_cast<const v16bf*>(&sK[(t * 16 + lr) * HD + ks * 32 + halfk]);
        sacc[t] = __builtin_amdgcn_wmma_f32_16x16x32_bf16(
            false, a, false, bb, (short)0, sacc[t], false, false);
      }
    // causal mask + scale
#pragma unroll
    for (int t = 0; t < 4; ++t)
#pragma unroll
      for (int e = 0; e < 8; ++e) {
        int rg = q0 + r0 + e + rowHalf;
        int cg = kb * 64 + t * 16 + lr;
        float s = sacc[t][e] * scale;
        sacc[t][e] = (cg <= rg) ? s : -1e30f;
      }
    // online softmax update per row (rows live in 16-lane halves)
#pragma unroll
    for (int e = 0; e < 8; ++e) {
      float v = fmaxf(fmaxf(sacc[0][e], sacc[1][e]), fmaxf(sacc[2][e], sacc[3][e]));
      v = fmaxf(v, __shfl_xor(v, 1)); v = fmaxf(v, __shfl_xor(v, 2));
      v = fmaxf(v, __shfl_xor(v, 4)); v = fmaxf(v, __shfl_xor(v, 8));
      float mnew = fmaxf(rm[e], v);
      float corr = __expf(rm[e] - mnew);
      rm[e] = mnew;
      rl[e] *= corr;
#pragma unroll
      for (int n = 0; n < 6; ++n) o[n][e] *= corr;
      float psum = 0.0f;
#pragma unroll
      for (int t = 0; t < 4; ++t) {
        float p = __expf(sacc[t][e] - mnew);
        sacc[t][e] = p;
        psum += p;
      }
      psum += __shfl_xor(psum, 1); psum += __shfl_xor(psum, 2);
      psum += __shfl_xor(psum, 4); psum += __shfl_xor(psum, 8);
      rl[e] += psum;
      int m = e + rowHalf;
#pragma unroll
      for (int t = 0; t < 4; ++t)
        sP[(r0 + m) * 64 + t * 16 + lr] = f2bf(sacc[t][e]);
    }
    __syncthreads();
    // O += P * V   (contract over 64 keys; V transposed in LDS)
#pragma unroll
    for (int ks = 0; ks < 2; ++ks) {
      v16bf a = *reinterpret_cast<const v16bf*>(&sP[(r0 + lr) * 64 + ks * 32 + halfk]);
#pragma unroll
      for (int n = 0; n < 6; ++n) {
        v16bf bb = *reinterpret_cast<const v16bf*>(&sVt[(n * 16 + lr) * 64 + ks * 32 + halfk]);
        o[n] = __builtin_amdgcn_wmma_f32_16x16x32_bf16(
            false, a, false, bb, (short)0, o[n], false, false);
      }
    }
  }
  // normalize + write AO (bf16)
#pragma unroll
  for (int e = 0; e < 8; ++e) {
    float inv = 1.0f / rl[e];
    int m = e + rowHalf;
    size_t row = (size_t)(b * SEQ + q0 + r0 + m) * (NH * VHD) + (size_t)h * VHD;
#pragma unroll
    for (int n = 0; n < 6; ++n)
      AO[row + n * 16 + lr] = f2bf(o[n][e] * inv);
  }
}

// ---------------------------------------------------------------------------
extern "C" void kernel_launch(void* const* d_in, const int* in_sizes, int n_in,
                              void* d_out, int out_size, void* d_ws, size_t ws_size,
                              hipStream_t stream) {
  (void)in_sizes; (void)n_in; (void)out_size; (void)ws_size;
  const float* hs  = (const float*)d_in[0];
  const int*   pid = (const int*)d_in[1];
  const float* Wq  = (const float*)d_in[2];
  const float* Wk  = (const float*)d_in[3];
  const float* Wv  = (const float*)d_in[4];
  const float* Wo  = (const float*)d_in[5];
  float* out = (float*)d_out;

  unsigned short* p   = (unsigned short*)d_ws;
  unsigned short* hsb = p; p += (size_t)BS * HIDDEN;
  unsigned short* wqb = p; p += (size_t)NH * HD * HIDDEN;
  unsigned short* wkb = p; p += (size_t)NKV * HD * HIDDEN;
  unsigned short* wvb = p; p += (size_t)NKV * VHD * HIDDEN;
  unsigned short* wob = p; p += (size_t)HIDDEN * NH * VHD;
  unsigned short* qb  = p; p += (size_t)BS * NH * HD;
  unsigned short* kbf = p; p += (size_t)BS * NKV * HD;
  unsigned short* vbf = p; p += (size_t)BS * NKV * VHD;
  unsigned short* aob = p; p += (size_t)BS * NH * VHD;

  auto cvt = [&](const float* s, unsigned short* d, size_t n) {
    int n4 = (int)(n >> 2);
    cvt_f32_bf16<<<(n4 + 255) / 256, 256, 0, stream>>>(s, d, n4);
  };
  cvt(hs, hsb, (size_t)BS * HIDDEN);
  cvt(Wq, wqb, (size_t)NH * HD * HIDDEN);
  cvt(Wk, wkb, (size_t)NKV * HD * HIDDEN);
  cvt(Wv, wvb, (size_t)NKV * VHD * HIDDEN);
  cvt(Wo, wob, (size_t)HIDDEN * NH * VHD);

  // QKV projections (bf16 out)
  gemm_bf16_wmma<false><<<dim3((NH * HD) / 128, BS / 128), 256, 0, stream>>>(
      hsb, wqb, qb, BS, NH * HD, HIDDEN);
  gemm_bf16_wmma<false><<<dim3((NKV * HD) / 128, BS / 128), 256, 0, stream>>>(
      hsb, wkb, kbf, BS, NKV * HD, HIDDEN);
  gemm_bf16_wmma<false><<<dim3((NKV * VHD) / 128, BS / 128), 256, 0, stream>>>(
      hsb, wvb, vbf, BS, NKV * VHD, HIDDEN);

  rope_kernel<<<(BS * (ROT / 2)) / 256, 256, 0, stream>>>(qb, kbf, pid);

  flash_attn_kernel<<<dim3(SEQ / 64, NH, BATCH), 128, 0, stream>>>(qb, kbf, vbf, aob);

  // output projection (fp32 out, straight into d_out)
  gemm_bf16_wmma<true><<<dim3(HIDDEN / 128, BS / 128), 256, 0, stream>>>(
      aob, wob, out, BS, HIDDEN, NH * VHD);
}
